// TransformerSparseAttention_57483842290325
// MI455X (gfx1250) — matrix operand order repaired and checked by reading, not compiled, not run on hardware
//
#include <hip/hip_runtime.h>

// Problem constants (from the reference)
constexpr int kB = 8;      // batch
constexpr int kS = 1024;   // sequence
constexpr int kE = 512;    // embed
constexpr int kH = 8;      // heads
constexpr int kD = 64;     // head dim
constexpr int kL = 4;      // layers
constexpr int kFIN = 128;  // input features
constexpr int kCLS = 10;
constexpr int kTOPK = 256; // S / sparsity_factor

typedef __attribute__((ext_vector_type(16))) __bf16 v16bf;
typedef __attribute__((ext_vector_type(8)))  __bf16 v8bf;
typedef __attribute__((ext_vector_type(8)))  float  v8f;

// float -> bf16, round-to-nearest-even
__device__ inline __bf16 f2bf(float f) {
  unsigned u = __builtin_bit_cast(unsigned, f);
  unsigned r = (u + 0x7FFFu + ((u >> 16) & 1u)) >> 16;
  unsigned short h = (unsigned short)r;
  return __builtin_bit_cast(__bf16, h);
}
__device__ inline float bf2f(__bf16 x) {
  unsigned short u = __builtin_bit_cast(unsigned short, x);
  return __builtin_bit_cast(float, ((unsigned)u) << 16);
}

// ---- WMMA bf16 operand loaders (16x16x32, wave32, ISA 7.12.2 layouts) ------
// A fragment: row fixed per lane; element e -> K = 8*half + (e<8 ? e : e+8).
__device__ inline v16bf ld_a(const __bf16* __restrict__ rowk0, int half) {
  v8bf lo = *(const v8bf*)(rowk0 + 8 * half);
  v8bf hi = *(const v8bf*)(rowk0 + 16 + 8 * half);
  return __builtin_shufflevector(lo, hi, 0, 1, 2, 3, 4, 5, 6, 7,
                                 8, 9, 10, 11, 12, 13, 14, 15);
}
// B fragment: N fixed per lane; element e -> K = 8*half + (e<8 ? e : e+8) (K-major storage).
__device__ inline v16bf ld_b(const __bf16* __restrict__ rowk0, int half) {
  const __bf16* p = rowk0 + 8 * half;
  v8bf lo = *(const v8bf*)(p);
  v8bf hi = *(const v8bf*)(p + 16);
  return __builtin_shufflevector(lo, hi, 0, 1, 2, 3, 4, 5, 6, 7,
                                 8, 9, 10, 11, 12, 13, 14, 15);
}

// ------------- GEMM: C[M,N] = A[M,K] @ Wt[N,K]^T + bias, bf16 in/out --------
// MODE 0: row-major bf16 C[M,N].
// MODE 1: V head-transposed store: C -> Vt[((b*H+h)*64+d)*S + s] (N==kE).
// One wave computes a 16x64 tile (A fragment reused across 4 B fragments).
template <int MODE>
__global__ __launch_bounds__(256) void gemm_bf16_kernel(
    const __bf16* __restrict__ A, const __bf16* __restrict__ Wt,
    const float* __restrict__ bias, __bf16* __restrict__ C,
    int M, int N, int K) {
  int lane = threadIdx.x & 31;
  int wave = threadIdx.x >> 5;
  int tile = blockIdx.x * 8 + wave;
  int tilesN = N >> 6;
  int tm = (tile / tilesN) << 4;
  int tn = (tile % tilesN) << 6;
  if (tm >= M) return;  // wave-uniform

  int half = (lane >> 4) & 1;
  int n = lane & 15;
  const __bf16* arow = A + (size_t)(tm + n) * K;  // row = lane&15 (both halves)
  const __bf16* brow0 = Wt + (size_t)(tn + n) * K;
  const __bf16* brow1 = brow0 + (size_t)16 * K;
  const __bf16* brow2 = brow0 + (size_t)32 * K;
  const __bf16* brow3 = brow0 + (size_t)48 * K;

  v8f acc0 = {}, acc1 = {}, acc2 = {}, acc3 = {};
  for (int k0 = 0; k0 < K; k0 += 32) {
    if (k0 + 32 < K) {
      __builtin_prefetch(arow + k0 + 32, 0, 0);
      __builtin_prefetch(brow0 + k0 + 32, 0, 0);
    }
    v16bf a = ld_a(arow + k0, half);
    v16bf b0 = ld_b(brow0 + k0, half);
    v16bf b1 = ld_b(brow1 + k0, half);
    v16bf b2 = ld_b(brow2 + k0, half);
    v16bf b3 = ld_b(brow3 + k0, half);
    acc0 = __builtin_amdgcn_wmma_f32_16x16x32_bf16(false, a, false, b0, (short)0, acc0, false, false);
    acc1 = __builtin_amdgcn_wmma_f32_16x16x32_bf16(false, a, false, b1, (short)0, acc1, false, false);
    acc2 = __builtin_amdgcn_wmma_f32_16x16x32_bf16(false, a, false, b2, (short)0, acc2, false, false);
    acc3 = __builtin_amdgcn_wmma_f32_16x16x32_bf16(false, a, false, b3, (short)0, acc3, false, false);
  }

#pragma unroll
  for (int j = 0; j < 4; ++j) {
    const v8f& acc = (j == 0) ? acc0 : (j == 1) ? acc1 : (j == 2) ? acc2 : acc3;
    int ng = tn + j * 16 + n;
    float bv = bias ? bias[ng] : 0.0f;
#pragma unroll
    for (int r = 0; r < 8; ++r) {
      int m = tm + r + 8 * half;  // C/D layout: VGPR r -> M = r + 8*half
      float val = acc[r] + bv;
      if (MODE == 0) {
        C[(size_t)m * N + ng] = f2bf(val);
      } else {
        int b = m >> 10, s = m & 1023;
        int hh = ng >> 6, d = ng & 63;
        C[(size_t)(((b * kH + hh) * kD + d) << 10) + s] = f2bf(val);
      }
    }
  }
}

// ------------- sparse attention: one block per (b, head, 16-query tile) -----
__global__ __launch_bounds__(128) void attn_kernel(
    const __bf16* __restrict__ Q, const __bf16* __restrict__ Kb,
    const __bf16* __restrict__ Vt, __bf16* __restrict__ O) {
  __shared__ float  sc[16][kS];      // 64 KB raw scores
  __shared__ __bf16 pb[16][kS];      // 32 KB probabilities (bf16)
  __shared__ unsigned hist[4][256];  // 4 KB radix histograms (one per wave)

  int lane = threadIdx.x & 31;
  int wave = threadIdx.x >> 5;  // 0..3
  int bid = blockIdx.x;
  int qb = bid & 63;
  int hh = (bid >> 6) & 7;
  int bb = bid >> 9;
  int half = (lane >> 4) & 1;
  int n = lane & 15;

  // ---- Phase 1: scores = (q @ k^T) / sqrt(d). A (q tile) hoisted. ----------
  const __bf16* qrow = Q + (size_t)(bb * kS + qb * 16 + n) * kE + hh * kD;
  v16bf a0 = ld_a(qrow + 0, half);
  v16bf a1 = ld_a(qrow + 32, half);
  for (int kt = wave * 16; kt < wave * 16 + 16; ++kt) {
    const __bf16* krow = Kb + (size_t)(bb * kS + kt * 16 + n) * kE + hh * kD;
    v16bf b0 = ld_b(krow + 0, half);
    v16bf b1 = ld_b(krow + 32, half);
    v8f acc = {};
    acc = __builtin_amdgcn_wmma_f32_16x16x32_bf16(false, a0, false, b0, (short)0, acc, false, false);
    acc = __builtin_amdgcn_wmma_f32_16x16x32_bf16(false, a1, false, b1, (short)0, acc, false, false);
#pragma unroll
    for (int r = 0; r < 8; ++r)
      sc[r + 8 * half][kt * 16 + n] = acc[r] * 0.125f;  // 1/sqrt(64)
  }
  __syncthreads();

  // ---- Phase 2: exact top-k via radix-256 select + masked softmax ----------
  // One row per wave at a time; every wave runs identical trip counts, so the
  // __syncthreads() below are block-uniform.
  for (int r = wave * 4; r < wave * 4 + 4; ++r) {
    unsigned u[32];
    float s[32];
#pragma unroll
    for (int i = 0; i < 32; ++i) {
      float f = sc[r][lane * 32 + i];
      s[i] = f;
      unsigned x = __builtin_bit_cast(unsigned, f);
      u[i] = (x & 0x80000000u) ? ~x : (x | 0x80000000u);  // order-preserving key
    }
    // Find t = k-th largest key (then keep u >= t: tie-inclusive semantics).
    unsigned prefix = 0;
    unsigned kneed = kTOPK;
    for (int pass = 0; pass < 4; ++pass) {
      int shift = 24 - 8 * pass;
#pragma unroll
      for (int j = 0; j < 8; ++j) hist[wave][lane * 8 + j] = 0;
      __syncthreads();
      unsigned himask = (pass == 0) ? 0u : (0xFFFFFFFFu << (shift + 8));
#pragma unroll
      for (int i = 0; i < 32; ++i) {
        if (((u[i] ^ prefix) & himask) == 0)
          atomicAdd(&hist[wave][(u[i] >> shift) & 255u], 1u);
      }
      __syncthreads();
      unsigned cnt[8], lsum = 0;
#pragma unroll
      for (int j = 0; j < 8; ++j) { cnt[j] = hist[wave][lane * 8 + j]; lsum += cnt[j]; }
      // suffix-inclusive scan across lanes (higher lane = higher-valued bins)
      unsigned run = lsum;
#pragma unroll
      for (int o = 1; o < 32; o <<= 1) {
        unsigned other = __shfl_down(run, o, 32);
        if (lane + o < 32) run += other;
      }
      unsigned above = run - lsum;  // count in bins strictly above my range
      int mybin = -1;
      unsigned myrem = kneed;
      if (above < kneed && above + lsum >= kneed) {  // crossing is in my bins
        unsigned acc2 = above;
#pragma unroll
        for (int j = 7; j >= 0; --j) {
          if (mybin < 0 && acc2 + cnt[j] >= kneed) { mybin = j; myrem = kneed - acc2; }
          acc2 += cnt[j];
        }
      }
      unsigned long long bal = __ballot(mybin >= 0);  // exactly one lane
      int src = (int)__builtin_ctzll(bal);
      int binj = __shfl(mybin, src, 32);
      unsigned rem = (unsigned)__shfl((int)myrem, src, 32);
      prefix |= (unsigned)(src * 8 + binj) << shift;
      kneed = rem;
    }
    unsigned t = prefix;

    // masked softmax over kept entries (kept max == row max)
    float m = -1e30f;
#pragma unroll
    for (int i = 0; i < 32; ++i)
      if (u[i] >= t) m = fmaxf(m, s[i]);
    for (int o = 16; o; o >>= 1) m = fmaxf(m, __shfl_xor(m, o, 32));
    float p[32];
    float sum = 0.0f;
#pragma unroll
    for (int i = 0; i < 32; ++i) {
      float e = (u[i] >= t) ? __expf(s[i] - m) : 0.0f;
      p[i] = e;
      sum += e;
    }
    for (int o = 16; o; o >>= 1) sum += __shfl_xor(sum, o, 32);
    float inv = 1.0f / sum;
#pragma unroll
    for (int i = 0; i < 32; ++i) pb[r][lane * 32 + i] = f2bf(p[i] * inv);
  }
  __syncthreads();

  // ---- Phase 3: O = P @ V.  Wave w owns output dims [16w, 16w+16). --------
  v8f acc = {};
  const __bf16* prow = &pb[n][0];  // A row = lane&15
  const __bf16* vrow = Vt + (size_t)((bb * kH + hh) * kD + wave * 16 + n) * kS;
  for (int k0 = 0; k0 < kS; k0 += 32) {
    v8bf alo = *(const v8bf*)(prow + k0 + 8 * half);
    v8bf ahi = *(const v8bf*)(prow + k0 + 8 * half + 16);
    v16bf a = __builtin_shufflevector(alo, ahi, 0, 1, 2, 3, 4, 5, 6, 7,
                                      8, 9, 10, 11, 12, 13, 14, 15);
    v16bf b = ld_b(vrow + k0, half);  // Vt rows are contiguous in keys
    acc = __builtin_amdgcn_wmma_f32_16x16x32_bf16(false, a, false, b, (short)0, acc, false, false);
  }
  __bf16* obase = O + (size_t)(bb * kS + qb * 16) * kE + hh * kD + wave * 16;
#pragma unroll
  for (int r = 0; r < 8; ++r)
    obase[(size_t)(r + 8 * half) * kE + n] = f2bf(acc[r]);
}

// ---------------- conversion / transpose passes (run once per launch) -------
__global__ void cvt_bf16_kernel(const float* __restrict__ in,
                                __bf16* __restrict__ out, int nelem) {
  int i = blockIdx.x * blockDim.x + threadIdx.x;
  if (i < nelem) out[i] = f2bf(in[i]);
}

// W[K,N] fp32 -> Wt[N,K] bf16
__global__ void transpose_cvt_kernel(const float* __restrict__ W,
                                     __bf16* __restrict__ Wt, int K, int N) {
  int i = blockIdx.x * blockDim.x + threadIdx.x;
  if (i >= K * N) return;
  int k = i / N, n = i % N;
  Wt[(size_t)n * K + k] = f2bf(W[i]);
}

// ---------------- head: mean over S (bf16 in, f32 out), then tiny FC --------
__global__ void mean_kernel(const __bf16* __restrict__ Hf, float* __restrict__ Hbar) {
  int idx = blockIdx.x * blockDim.x + threadIdx.x;  // over B * (E/8)
  if (idx >= kB * (kE / 8)) return;
  int b = idx / (kE / 8), c8 = idx % (kE / 8);
  const __bf16* base = Hf + (size_t)b * kS * kE + c8 * 8;
  float acc[8] = {};
  for (int t = 0; t < kS; ++t) {
    v8bf v = *(const v8bf*)(base + (size_t)t * kE);
#pragma unroll
    for (int j = 0; j < 8; ++j) acc[j] += bf2f(v[j]);
  }
#pragma unroll
  for (int j = 0; j < 8; ++j) Hbar[b * kE + c8 * 8 + j] = acc[j] * (1.0f / kS);
}

__global__ void fc_kernel(const float* __restrict__ Hbar, const float* __restrict__ Wf,
                          const float* __restrict__ bf, float* __restrict__ out) {
  int idx = blockIdx.x * blockDim.x + threadIdx.x;
  if (idx >= kB * kCLS) return;
  int b = idx / kCLS, c = idx % kCLS;
  float s = bf[c];
  for (int e = 0; e < kE; ++e) s += Hbar[b * kE + e] * Wf[e * kCLS + c];
  out[idx] = s;
}

// ---------------------------------------------------------------------------
extern "C" void kernel_launch(void* const* d_in, const int* in_sizes, int n_in,
                              void* d_out, int out_size, void* d_ws, size_t ws_size,
                              hipStream_t stream) {
  (void)in_sizes; (void)n_in; (void)out_size; (void)ws_size;
  const float* x    = (const float*)d_in[0];
  const float* embw = (const float*)d_in[1];
  const float* embb = (const float*)d_in[2];
  const float* Wq   = (const float*)d_in[3];
  const float* Wk   = (const float*)d_in[4];
  const float* Wv   = (const float*)d_in[5];
  const float* Wo   = (const float*)d_in[6];
  const float* bq   = (const float*)d_in[7];
  const float* bk   = (const float*)d_in[8];
  const float* bv   = (const float*)d_in[9];
  const float* bo   = (const float*)d_in[10];
  const float* fcw  = (const float*)d_in[11];
  const float* fcb  = (const float*)d_in[12];
  float* out = (float*)d_out;

  const size_t act = (size_t)kB * kS * kE;  // 4M elems (bf16 => 8 MB)
  char* p = (char*)d_ws;
  auto grab = [&](size_t bytes) { char* r = p; p += (bytes + 255) & ~(size_t)255; return r; };
  __bf16* xb    = (__bf16*)grab((size_t)kB * kS * kFIN * 2);
  __bf16* embwt = (__bf16*)grab((size_t)kFIN * kE * 2);
  __bf16* wqt   = (__bf16*)grab((size_t)kL * kE * kE * 2);
  __bf16* wkt   = (__bf16*)grab((size_t)kL * kE * kE * 2);
  __bf16* wvt   = (__bf16*)grab((size_t)kL * kE * kE * 2);
  __bf16* wot   = (__bf16*)grab((size_t)kL * kE * kE * 2);
  __bf16* h0    = (__bf16*)grab(act * 2);
  __bf16* h1    = (__bf16*)grab(act * 2);
  __bf16* q     = (__bf16*)grab(act * 2);
  __bf16* kk    = (__bf16*)grab(act * 2);
  __bf16* vt    = (__bf16*)grab(act * 2);  // [B][H][64][S]
  __bf16* ao    = (__bf16*)grab(act * 2);
  float*  hbar  = (float*)grab((size_t)kB * kE * 4);

  dim3 cblk(256);
  // input + weight conversion (bandwidth-trivial, once per launch)
  {
    int nx = kB * kS * kFIN;
    cvt_bf16_kernel<<<dim3((nx + 255) / 256), cblk, 0, stream>>>(x, xb, nx);
    int ne = kFIN * kE;
    transpose_cvt_kernel<<<dim3((ne + 255) / 256), cblk, 0, stream>>>(embw, embwt, kFIN, kE);
    int nw = kE * kE;
    dim3 wgrid((nw + 255) / 256);
    for (int l = 0; l < kL; ++l) {
      size_t o = (size_t)l * kE * kE;
      transpose_cvt_kernel<<<wgrid, cblk, 0, stream>>>(Wq + o, wqt + o, kE, kE);
      transpose_cvt_kernel<<<wgrid, cblk, 0, stream>>>(Wk + o, wkt + o, kE, kE);
      transpose_cvt_kernel<<<wgrid, cblk, 0, stream>>>(Wv + o, wvt + o, kE, kE);
      transpose_cvt_kernel<<<wgrid, cblk, 0, stream>>>(Wo + o, wot + o, kE, kE);
    }
  }

  const int M = kB * kS;                        // 8192
  const int tiles = (M / 16) * (kE / 64);       // 4096 (16x64 per wave)
  dim3 gblk(256), ggrid(tiles / 8);
  dim3 ablk(128), agrid(kB * kH * (kS / 16));   // 4096

  // embedding: h0 = x @ emb_w + emb_b
  gemm_bf16_kernel<0><<<ggrid, gblk, 0, stream>>>(xb, embwt, embb, h0, M, kE, kFIN);

  __bf16* hin = h0; __bf16* hout = h1;
  for (int l = 0; l < kL; ++l) {
    const size_t wo = (size_t)l * kE * kE;
    gemm_bf16_kernel<0><<<ggrid, gblk, 0, stream>>>(hin, wqt + wo, bq + l * kE, q,  M, kE, kE);
    gemm_bf16_kernel<0><<<ggrid, gblk, 0, stream>>>(hin, wkt + wo, bk + l * kE, kk, M, kE, kE);
    gemm_bf16_kernel<1><<<ggrid, gblk, 0, stream>>>(hin, wvt + wo, bv + l * kE, vt, M, kE, kE);
    attn_kernel<<<agrid, ablk, 0, stream>>>(q, kk, vt, ao);
    gemm_bf16_kernel<0><<<ggrid, gblk, 0, stream>>>(ao, wot + wo, bo + l * kE, hout, M, kE, kE);
    __bf16* t = hin; hin = hout; hout = t;
  }

  mean_kernel<<<dim3((kB * (kE / 8) + 255) / 256), gblk, 0, stream>>>(hin, hbar);
  fc_kernel<<<dim3(1), ablk, 0, stream>>>(hbar, fcw, fcb, out);
}
